// MultiHeadAttention_42923903156956
// MI455X (gfx1250) — matrix operand order repaired and checked by reading, not compiled, not run on hardware
//
#include <hip/hip_runtime.h>

#define N_STATE   1024
#define N_HEAD    16
#define HEAD_DIM  64
#define BATCH     8
#define SEQ       1500
#define SEQ_PAD   1504            // V^T row stride
#define MTOT      (BATCH * SEQ)   // 12000, multiple of 16

typedef __attribute__((ext_vector_type(16))) __bf16          v16bf;
typedef __attribute__((ext_vector_type(8)))  float           v8f;
typedef __attribute__((ext_vector_type(8)))  unsigned short  su8;     // 16 bytes
typedef __attribute__((ext_vector_type(4)))  unsigned int    uint32x4;
typedef __attribute__((ext_vector_type(4)))  int             int32x4;
typedef __attribute__((ext_vector_type(8)))  int             int32x8;

union Frag16 {
    v16bf          v;
    su8            h[2];
    unsigned short u[16];
};

__device__ __forceinline__ unsigned short f2bf(float f) {
    unsigned int x = __float_as_uint(f);
    unsigned int r = x + 0x7FFFu + ((x >> 16) & 1u);   // round-to-nearest-even
    return (unsigned short)(r >> 16);
}

__device__ __forceinline__ v8f v8f_zero() {
    v8f r;
    #pragma unroll
    for (int i = 0; i < 8; ++i) r[i] = 0.0f;
    return r;
}

__device__ __forceinline__ v8f wmma_bf16(const Frag16& a, const Frag16& b, v8f c) {
    return __builtin_amdgcn_wmma_f32_16x16x32_bf16(false, a.v, false, b.v,
                                                   (short)0, c, false, false);
}

// ---------------------------------------------------------------------------
// Tensor Data Mover: 2D tile (global -> LDS), bf16 elements.
// D# layout per CDNA5 ISA.  OOB rows/cols read as zero.
// ---------------------------------------------------------------------------
__device__ __forceinline__ void tdm_load_2d(unsigned lds_off, const void* gaddr,
                                            unsigned tile_d0, unsigned tile_d1,
                                            unsigned tens_d0, unsigned tens_d1,
                                            unsigned stride0) {
    unsigned long long ga = (unsigned long long)(size_t)gaddr;
    uint32x4 g0;
    g0[0] = 1u;                                        // count=1, user descriptor
    g0[1] = lds_off;                                   // lds_addr (bytes)
    g0[2] = (unsigned)(ga & 0xFFFFFFFFu);              // global_addr[31:0]
    g0[3] = (unsigned)((ga >> 32) & 0x1FFFFFFu)        // global_addr[56:32]
          | (2u << 30);                                // type = 2 ("image")
    int32x8 g1;
    g1[0] = (int)(1u << 16);                           // data_size=1 (2 bytes)
    g1[1] = (int)(tens_d0 << 16);                      // tensor_dim0[15:0]
    g1[2] = (int)((tens_d0 >> 16) | (tens_d1 << 16));  // dim0 hi | dim1 lo
    g1[3] = (int)((tens_d1 >> 16) | (tile_d0 << 16));  // dim1 hi | tile_dim0
    g1[4] = (int)(tile_d1 & 0xFFFFu);                  // tile_dim1 (tile_dim2=0)
    g1[5] = (int)stride0;                              // tensor_dim0_stride lo
    g1[6] = 0;                                         // stride0 hi | dim1_stride lo
    g1[7] = 0;
    int32x4 z4;
    z4[0] = 0; z4[1] = 0; z4[2] = 0; z4[3] = 0;
#if defined(__clang_major__) && (__clang_major__ >= 23)
    int32x8 z8;
    #pragma unroll
    for (int i = 0; i < 8; ++i) z8[i] = 0;
    __builtin_amdgcn_tensor_load_to_lds(g0, g1, z4, z4, z8, 0);
#else
    __builtin_amdgcn_tensor_load_to_lds(g0, g1, z4, z4, 0);
#endif
}

__device__ __forceinline__ unsigned lds_byte_off(const void* p) {
    return (unsigned)(size_t)p;                        // low 32 bits = LDS offset
}

// ---------------------------------------------------------------------------
// fp32 -> bf16 elementwise
// ---------------------------------------------------------------------------
__global__ void k_f32_to_bf16(const float* __restrict__ src,
                              unsigned short* __restrict__ dst, int n) {
    int i = blockIdx.x * blockDim.x + threadIdx.x;
    if (i < n) dst[i] = f2bf(src[i]);
}

__global__ void k_zero_f32(float* __restrict__ p, int n) {
    int i = blockIdx.x * blockDim.x + threadIdx.x;
    if (i < n) p[i] = 0.0f;
}

// ---------------------------------------------------------------------------
// Pack W [N=1024,K=1024] fp32 (torch Linear layout) into WMMA B-fragment
// layout fused with transpose + bf16 conversion (see round-2 notes).
// ---------------------------------------------------------------------------
__global__ void k_pack_w(const float* __restrict__ src,
                         unsigned short* __restrict__ dst) {
    int t    = blockIdx.x * blockDim.x + threadIdx.x;  // 0 .. 65535
    int lane = t & 31;
    int n16  = (t >> 5) & 63;
    int kt   = t >> 11;                                // 0..31
    int n    = lane & 15;
    int half = lane >> 4;
    const float* s = src + (size_t)(n16 * 16 + n) * 1024 + kt * 32 + 16 * half;
    unsigned short* d = dst + (size_t)t * 16;
    #pragma unroll
    for (int j = 0; j < 16; ++j) d[j] = f2bf(s[j]);
}

// ---------------------------------------------------------------------------
// QKV projection: C[16x64] per wave.  VT=0: out [B,H,S,D]; VT=1: out [B,H,D,S_pad].
// ---------------------------------------------------------------------------
template <int VT>
__global__ void __launch_bounds__(32)
k_gemm_qkv(const unsigned short* __restrict__ xb,
           const unsigned short* __restrict__ wpk,
           const float* __restrict__ bias,
           unsigned short* __restrict__ dst)
{
    const int lane = threadIdx.x;
    const int n    = lane & 15;
    const int half = lane >> 4;
    const int mt   = blockIdx.x;
    const int nt   = blockIdx.y;

    v8f acc[4];
    #pragma unroll
    for (int i = 0; i < 4; ++i) acc[i] = v8f_zero();

    const unsigned short* arow = xb + (size_t)(mt * 16 + n) * 1024;

    for (int kt = 0; kt < 32; ++kt) {
        Frag16 a;
        a.h[0] = *(const su8*)(arow + kt * 32 + 16 * half);
        a.h[1] = *(const su8*)(arow + kt * 32 + 16 * half + 8);
        #pragma unroll
        for (int dt = 0; dt < 4; ++dt) {
            const unsigned short* bp =
                wpk + ((size_t)((kt * 64) + (nt * 4 + dt)) * 32 + lane) * 16;
            Frag16 b;
            b.h[0] = *(const su8*)(bp);
            b.h[1] = *(const su8*)(bp + 8);
            acc[dt] = wmma_bf16(a, b, acc[dt]);
        }
    }

    float bvals[4];
    #pragma unroll
    for (int dt = 0; dt < 4; ++dt)
        bvals[dt] = bias[nt * 64 + dt * 16 + n];

    #pragma unroll
    for (int r = 0; r < 8; ++r) {
        int row  = mt * 16 + 8 * half + r;
        int bidx = row / SEQ;
        int s    = row - bidx * SEQ;
        #pragma unroll
        for (int dt = 0; dt < 4; ++dt) {
            int col = nt * 64 + dt * 16 + n;
            int hh  = col >> 6;
            int d   = col & 63;
            unsigned short val = f2bf(acc[dt][r] + bvals[dt]);
            if (VT == 0)
                dst[((size_t)(bidx * 16 + hh) * SEQ + s) * 64 + d] = val;
            else
                dst[((size_t)(bidx * 16 + hh) * 64 + d) * SEQ_PAD + s] = val;
        }
    }
}

// ---------------------------------------------------------------------------
// Causal flash attention.  4 waves / workgroup = 64 queries of one (b,h).
// K tile (32x64) and V^T tile (64x32) staged in LDS by the Tensor Data Mover,
// double-buffered; synchronized with s_wait_tensorcnt + workgroup barrier.
// Q,K: [B,H,S,D] bf16.  V: [B,H,D,S_pad] bf16.  O: [12000,1024] bf16.
// ---------------------------------------------------------------------------
__global__ void __launch_bounds__(128)
k_attention(const unsigned short* __restrict__ Q,
            const unsigned short* __restrict__ K,
            const unsigned short* __restrict__ V,
            unsigned short* __restrict__ O)
{
    __shared__ __align__(16) unsigned short Kt[2][32 * 64];   // [key][dim]
    __shared__ __align__(16) unsigned short Vt[2][64 * 32];   // [dim][key]
    __shared__ __align__(16) unsigned short Pl[4][16 * 32];   // per-wave P tile

    const int tid  = threadIdx.x;
    const int lane = tid & 31;
    const int wid  = tid >> 5;
    const int n    = lane & 15;
    const int half = lane >> 4;
    const int h    = blockIdx.y;
    const int b    = blockIdx.z;
    const int q0   = blockIdx.x * 64 + wid * 16;

    const unsigned short* Qh = Q + (size_t)(b * 16 + h) * SEQ * 64;
    const unsigned short* Kh = K + (size_t)(b * 16 + h) * SEQ * 64;
    const unsigned short* Vh = V + (size_t)(b * 16 + h) * 64 * SEQ_PAD;

    // Q A-fragments (dims 0..31 / 32..63); row = lane&15, clamped
    int qrowA = q0 + n;
    if (qrowA > SEQ - 1) qrowA = SEQ - 1;
    Frag16 aQ[2];
    #pragma unroll
    for (int s = 0; s < 2; ++s) {
        const unsigned short* p = Qh + (size_t)qrowA * 64 + 32 * s;
        aQ[s].h[0] = *(const su8*)(p + 16 * half);
        aQ[s].h[1] = *(const su8*)(p + 16 * half + 8);
    }

    v8f o[4];
    #pragma unroll
    for (int i = 0; i < 4; ++i) o[i] = v8f_zero();
    float mrow[8], lrow[8];
    #pragma unroll
    for (int r = 0; r < 8; ++r) { mrow[r] = -3.0e38f; lrow[r] = 0.0f; }

    const int kend = ((blockIdx.x * 64 + 64) < SEQ) ? (blockIdx.x * 64 + 64) : SEQ;

    // prologue: stage tile 0 into buffer 0
    if (wid == 0) {
        tdm_load_2d(lds_byte_off(&Kt[0][0]), Kh, 64, 32, 64, SEQ, 64);
        tdm_load_2d(lds_byte_off(&Vt[0][0]), Vh, 32, 64, SEQ, 64, SEQ_PAD);
        __builtin_amdgcn_s_wait_tensorcnt(0);
    }
    __syncthreads();

    int buf = 0;
    for (int k0 = 0; k0 < kend; k0 += 32) {
        const int k1 = k0 + 32;
        const bool more = (k1 < kend);
        if (more && wid == 0) {                 // async prefetch of next tile
            tdm_load_2d(lds_byte_off(&Kt[buf ^ 1][0]),
                        Kh + (size_t)k1 * 64, 64, 32, 64, (unsigned)(SEQ - k1), 64);
            tdm_load_2d(lds_byte_off(&Vt[buf ^ 1][0]),
                        Vh + k1, 32, 64, (unsigned)(SEQ - k1), 64, SEQ_PAD);
        }

        const unsigned short* Kl = &Kt[buf][0];
        const unsigned short* Vl = &Vt[buf][0];

        // ---- scores: two 16x16 tiles over 32 keys ----
        v8f sc[2];
        #pragma unroll
        for (int sub = 0; sub < 2; ++sub) {
            v8f c = v8f_zero();
            #pragma unroll
            for (int s = 0; s < 2; ++s) {
                Frag16 bK;
                const unsigned short* p = Kl + (sub * 16 + n) * 64 + 32 * s + 16 * half;
                bK.h[0] = *(const su8*)(p);
                bK.h[1] = *(const su8*)(p + 8);
                c = wmma_bf16(aQ[s], bK, c);
            }
            sc[sub] = c;
        }

        // ---- scale + causal/length mask ----
        #pragma unroll
        for (int sub = 0; sub < 2; ++sub) {
            int key = k0 + sub * 16 + n;
            #pragma unroll
            for (int r = 0; r < 8; ++r) {
                int q = q0 + 8 * half + r;
                float s = sc[sub][r] * 0.125f;
                if (key > q || key >= SEQ) s = -1.0e30f;
                sc[sub][r] = s;
            }
        }

        // ---- online softmax (row reduction across the 16-lane group) ----
        #pragma unroll
        for (int r = 0; r < 8; ++r) {
            float t = fmaxf(sc[0][r], sc[1][r]);
            #pragma unroll
            for (int sh = 1; sh < 16; sh <<= 1)
                t = fmaxf(t, __shfl_xor(t, sh, 32));
            float mnew  = fmaxf(mrow[r], t);
            float scale = __expf(mrow[r] - mnew);
            float p0 = __expf(sc[0][r] - mnew);
            float p1 = __expf(sc[1][r] - mnew);
            sc[0][r] = p0;
            sc[1][r] = p1;
            float rs = p0 + p1;
            #pragma unroll
            for (int sh = 1; sh < 16; sh <<= 1)
                rs += __shfl_xor(rs, sh, 32);
            lrow[r] = lrow[r] * scale + rs;
            mrow[r] = mnew;
            #pragma unroll
            for (int dt = 0; dt < 4; ++dt) o[dt][r] *= scale;
        }

        // ---- stage P (16x32) in per-wave LDS: C-layout -> A-layout ----
        #pragma unroll
        for (int sub = 0; sub < 2; ++sub)
            #pragma unroll
            for (int r = 0; r < 8; ++r)
                Pl[wid][(8 * half + r) * 32 + sub * 16 + n] = f2bf(sc[sub][r]);
        asm volatile("s_wait_dscnt 0" ::: "memory");   // intra-wave DS order
        Frag16 aP;
        {
            const unsigned short* pr = &Pl[wid][n * 32];
            aP.h[0] = *(const su8*)(pr + 16 * half);
            aP.h[1] = *(const su8*)(pr + 16 * half + 8);
        }
        asm volatile("" ::: "memory");

        // ---- O += P * V ----
        #pragma unroll
        for (int dt = 0; dt < 4; ++dt) {
            const unsigned short* vp = Vl + (dt * 16 + n) * 32 + 16 * half;
            Frag16 bV;
            bV.h[0] = *(const su8*)(vp);
            bV.h[1] = *(const su8*)(vp + 8);
            o[dt] = wmma_bf16(aP, bV, o[dt]);
        }

        if (more && wid == 0) __builtin_amdgcn_s_wait_tensorcnt(0);
        __syncthreads();                                // tile consumed / next ready
        buf ^= 1;
    }

    // ---- normalize + store (concat heads: col = h*64 + d) ----
    #pragma unroll
    for (int r = 0; r < 8; ++r) {
        int q = q0 + 8 * half + r;
        if (q >= SEQ) continue;
        float inv = 1.0f / lrow[r];
        size_t row = (size_t)(b * SEQ + q) * 1024;
        #pragma unroll
        for (int dt = 0; dt < 4; ++dt)
            O[row + h * 64 + dt * 16 + n] = f2bf(o[dt][r] * inv);
    }
}

// ---------------------------------------------------------------------------
// Output projection: out = A(bf16 [12000,1024]) @ Wo^T + bo, fp32 stores.
// ---------------------------------------------------------------------------
__global__ void __launch_bounds__(32)
k_gemm_out(const unsigned short* __restrict__ A,
           const unsigned short* __restrict__ wpk,
           const float* __restrict__ bias,
           float* __restrict__ out)
{
    const int lane = threadIdx.x;
    const int n    = lane & 15;
    const int half = lane >> 4;
    const int mt   = blockIdx.x;
    const int nt   = blockIdx.y;

    v8f acc[4];
    #pragma unroll
    for (int i = 0; i < 4; ++i) acc[i] = v8f_zero();

    const unsigned short* arow = A + (size_t)(mt * 16 + n) * 1024;

    for (int kt = 0; kt < 32; ++kt) {
        Frag16 a;
        a.h[0] = *(const su8*)(arow + kt * 32 + 16 * half);
        a.h[1] = *(const su8*)(arow + kt * 32 + 16 * half + 8);
        #pragma unroll
        for (int dt = 0; dt < 4; ++dt) {
            const unsigned short* bp =
                wpk + ((size_t)((kt * 64) + (nt * 4 + dt)) * 32 + lane) * 16;
            Frag16 b;
            b.h[0] = *(const su8*)(bp);
            b.h[1] = *(const su8*)(bp + 8);
            acc[dt] = wmma_bf16(a, b, acc[dt]);
        }
    }

    #pragma unroll
    for (int r = 0; r < 8; ++r) {
        size_t row = (size_t)(mt * 16 + 8 * half + r) * 1024;
        #pragma unroll
        for (int dt = 0; dt < 4; ++dt) {
            int col = nt * 64 + dt * 16 + n;
            out[row + col] = acc[dt][r] + bias[col];
        }
    }
}

// ---------------------------------------------------------------------------
// Workspace layout (bytes)
// ---------------------------------------------------------------------------
static const size_t OFF_XB  = 0;                       // 12000*1024 bf16
static const size_t OFF_WQP = 24576000;                // packed 1024*1024 bf16
static const size_t OFF_WKP = 26673152;
static const size_t OFF_WVP = 28770304;
static const size_t OFF_WOP = 30867456;
static const size_t OFF_Q   = 32964608;                // [B,H,S,D] bf16
static const size_t OFF_K   = 57540608;                // [B,H,S,D] bf16
static const size_t OFF_VT  = 82116608;                // [B,H,D,S_pad] bf16
static const size_t OFF_AO  = 106758144;               // [12000,1024] bf16
static const size_t OFF_ZB  = 131334144;               // 1024 fp32 zeros
// end = 131338240 bytes

extern "C" void kernel_launch(void* const* d_in, const int* in_sizes, int n_in,
                              void* d_out, int out_size, void* d_ws, size_t ws_size,
                              hipStream_t stream) {
    (void)in_sizes; (void)n_in; (void)out_size; (void)ws_size;

    const float* x    = (const float*)d_in[0];
    // d_in[1] is the additive causal mask; causality is applied analytically.
    const float* Wq   = (const float*)d_in[2];
    const float* bq   = (const float*)d_in[3];
    const float* Wk   = (const float*)d_in[4];
    const float* Wv   = (const float*)d_in[5];
    const float* bv   = (const float*)d_in[6];
    const float* Wo   = (const float*)d_in[7];
    const float* bo   = (const float*)d_in[8];
    float*       out  = (float*)d_out;

    char* ws = (char*)d_ws;
    unsigned short* xb  = (unsigned short*)(ws + OFF_XB);
    unsigned short* wqp = (unsigned short*)(ws + OFF_WQP);
    unsigned short* wkp = (unsigned short*)(ws + OFF_WKP);
    unsigned short* wvp = (unsigned short*)(ws + OFF_WVP);
    unsigned short* wop = (unsigned short*)(ws + OFF_WOP);
    unsigned short* Qb  = (unsigned short*)(ws + OFF_Q);
    unsigned short* Kb  = (unsigned short*)(ws + OFF_K);
    unsigned short* Vt  = (unsigned short*)(ws + OFF_VT);
    unsigned short* aO  = (unsigned short*)(ws + OFF_AO);
    float*          zb  = (float*)(ws + OFF_ZB);

    const int nx = MTOT * N_STATE;                     // 12,288,000
    k_f32_to_bf16<<<(nx + 255) / 256, 256, 0, stream>>>(x, xb, nx);

    k_pack_w<<<256, 256, 0, stream>>>(Wq, wqp);
    k_pack_w<<<256, 256, 0, stream>>>(Wk, wkp);
    k_pack_w<<<256, 256, 0, stream>>>(Wv, wvp);
    k_pack_w<<<256, 256, 0, stream>>>(Wo, wop);

    k_zero_f32<<<4, 256, 0, stream>>>(zb, 1024);

    dim3 gg(MTOT / 16, N_STATE / 64);                  // 750 x 16
    k_gemm_qkv<0><<<gg, 32, 0, stream>>>(xb, wqp, bq, Qb);
    k_gemm_qkv<0><<<gg, 32, 0, stream>>>(xb, wkp, zb, Kb);
    k_gemm_qkv<1><<<gg, 32, 0, stream>>>(xb, wvp, bv, Vt);

    dim3 ga((SEQ + 63) / 64, N_HEAD, BATCH);           // 24 x 16 x 8, 128 thr
    k_attention<<<ga, 128, 0, stream>>>(Qb, Kb, Vt, aO);

    k_gemm_out<<<gg, 32, 0, stream>>>(aO, wop, bo, out);
}
